// ConfidenceGatedGAT_4337916969236
// MI455X (gfx1250) — matrix operand order repaired and checked by reading, not compiled, not run on hardware
//
#include <hip/hip_runtime.h>
#include <hip/hip_bf16.h>

typedef __attribute__((ext_vector_type(16))) _Float16 v16h;
typedef __attribute__((ext_vector_type(8)))  _Float16 v8h;
typedef __attribute__((ext_vector_type(8)))  float    v8f;

#define GN 20000
#define GE 640000
#define GIN 256
#define GC 64
#define GHC 256
#define NEG_SLOPE 0.2f

__device__ __forceinline__ v8h cvt8(const float4 a, const float4 b) {
    v8h r;
    r[0] = (_Float16)a.x; r[1] = (_Float16)a.y; r[2] = (_Float16)a.z; r[3] = (_Float16)a.w;
    r[4] = (_Float16)b.x; r[5] = (_Float16)b.y; r[6] = (_Float16)b.z; r[7] = (_Float16)b.w;
    return r;
}

// ---------------------------------------------------------------------------
// fp32-in / f16-WMMA / fp32-out GEMM:  C[N x M] = act(A[N x K] @ B[K x M] + bias)
// 256 threads = 8 waves; block tile 64 rows x 32 cols; K-tile 64 (2 WMMAs per
// barrier interval). A staged row-major [row][k]; B staged TRANSPOSED [col][k]
// so every lane's v16h fragment is two contiguous ds_load_b128 per ISA 7.12.2.
// K is always a multiple of 64 here (64 or 256).
// ---------------------------------------------------------------------------
__global__ __launch_bounds__(256)
void gemm_wmma_kernel(const float* __restrict__ A, const float* __restrict__ B,
                      const float* __restrict__ bias, float* __restrict__ C,
                      int N, int K, int M, int do_relu)
{
    __shared__ _Float16 As[64][72];   // 64 rows x 64 k  (+8 pad -> 144B stride, 16B aligned)
    __shared__ _Float16 Bs[32][72];   // 32 cols x 64 k  (transposed), same stride

    const int tid  = threadIdx.x;
    const int wave = tid >> 5;
    const int lane = tid & 31;
    const int row0 = blockIdx.y * 64;
    const int col0 = blockIdx.x * 32;
    const int twr  = wave & 3;        // wave tile row (0..3)
    const int twc  = wave >> 2;       // wave tile col (0..1)
    const int m    = lane & 15;
    const int half = lane >> 4;

    // A staging: thread -> (row, 16 consecutive k)  [branchless row guard]
    const int ar   = tid >> 2;                 // 0..63
    const int ac   = (tid & 3) * 16;           // 0,16,32,48
    const int agr  = row0 + ar;
    const int agrc = (agr < N) ? agr : (N - 1);
    const _Float16 ascl = (_Float16)((agr < N) ? 1.f : 0.f);

    // B staging: thread -> (k, 8 consecutive cols); no guards (M,K multiples of 32/64)
    const int bk = tid >> 2;                   // 0..63
    const int bc = (tid & 3) * 8;              // 0,8,16,24

    v8f acc = {};

    for (int kk = 0; kk < K; kk += 64) {
        const float* ap = A + (size_t)agrc * K + kk + ac;
        if (kk + 64 < K) __builtin_prefetch(ap + 64, 0, 3);

        // ---- stage A: global_load_b128 x4 -> ds_store_b128 x2
        float4 a0 = ((const float4*)ap)[0];
        float4 a1 = ((const float4*)ap)[1];
        float4 a2 = ((const float4*)ap)[2];
        float4 a3 = ((const float4*)ap)[3];
        v8h alo = cvt8(a0, a1) * ascl;
        v8h ahi = cvt8(a2, a3) * ascl;
        *(v8h*)&As[ar][ac]     = alo;
        *(v8h*)&As[ar][ac + 8] = ahi;

        // ---- stage B transposed: global_load_b128 x2 -> packed ds_store_b16
        const float* bp = B + (size_t)(kk + bk) * M + col0 + bc;
        float4 b0 = ((const float4*)bp)[0];
        float4 b1 = ((const float4*)bp)[1];
        Bs[bc + 0][bk] = (_Float16)b0.x;
        Bs[bc + 1][bk] = (_Float16)b0.y;
        Bs[bc + 2][bk] = (_Float16)b0.z;
        Bs[bc + 3][bk] = (_Float16)b0.w;
        Bs[bc + 4][bk] = (_Float16)b1.x;
        Bs[bc + 5][bk] = (_Float16)b1.y;
        Bs[bc + 6][bk] = (_Float16)b1.z;
        Bs[bc + 7][bk] = (_Float16)b1.w;
        __syncthreads();

        // ---- two K=32 WMMA steps, fragments via contiguous ds_load_b128
#pragma unroll
        for (int j = 0; j < 2; ++j) {
            // A 16x32 f16 (ISA 7.12.2): lane (m,half) holds K runs
            //   [half*8 .. half*8+7] and [16+half*8 .. 16+half*8+7]
            const v8h fa0 = *(const v8h*)&As[twr * 16 + m][j * 32 + half * 8];
            const v8h fa1 = *(const v8h*)&As[twr * 16 + m][j * 32 + 16 + half * 8];
            // B 32x16 f16: lane (n=m,half) holds K runs [half*8 .. half*8+7] and [16+half*8 .. 16+half*8+7]
            const v8h fb0 = *(const v8h*)&Bs[twc * 16 + m][j * 32 + half * 8];
            const v8h fb1 = *(const v8h*)&Bs[twc * 16 + m][j * 32 + 16 + half * 8];
            v16h af = __builtin_shufflevector(fa0, fa1, 0,1,2,3,4,5,6,7,8,9,10,11,12,13,14,15);
            v16h bf = __builtin_shufflevector(fb0, fb1, 0,1,2,3,4,5,6,7,8,9,10,11,12,13,14,15);
            acc = __builtin_amdgcn_wmma_f32_16x16x32_f16(false, af, false, bf,
                                                         (short)0, acc, false, false);
        }
        __syncthreads();
    }

    // Epilogue. D layout: VGPR r, lanes 0-15 -> row r, lanes 16-31 -> row 8+r;
    // col = lane&15. Bias is loop-invariant per lane: load it ONCE before the
    // 8-row write loop (avoids per-row branch + load + wait).
    const int gc  = col0 + twc * 16 + m;
    const int gr0 = row0 + twr * 16 + half * 8;
    const float bv = bias ? bias[gc] : 0.f;
#pragma unroll
    for (int r = 0; r < 8; ++r) {
        int gr = gr0 + r;
        if (gr < N) {
            float v = acc[r] + bv;
            if (do_relu) v = fmaxf(v, 0.f);
            C[(size_t)gr * M + gc] = v;
        }
    }
}

// ---------------------------------------------------------------------------
// Per-node attention scores: s_src[n,h] = sum_c hl[n,h,c]*a_src[h,c]
// ---------------------------------------------------------------------------
__global__ __launch_bounds__(256)
void attn_score_kernel(const float* __restrict__ hl,
                       const float* __restrict__ a_src, const float* __restrict__ a_dst,
                       float* __restrict__ s_src, float* __restrict__ s_dst, int N)
{
    __shared__ float red[256];
    int n = blockIdx.x;
    if (n >= N) return;
    int t = threadIdx.x;
    int hh = t >> 6, c = t & 63;
    float hv = hl[(size_t)n * GHC + t];

    red[t] = hv * a_src[t];
    __syncthreads();
    for (int s = 32; s > 0; s >>= 1) { if (c < s) red[t] += red[t + s]; __syncthreads(); }
    if (c == 0) s_src[n * 4 + hh] = red[t];
    __syncthreads();

    red[t] = hv * a_dst[t];
    __syncthreads();
    for (int s = 32; s > 0; s >>= 1) { if (c < s) red[t] += red[t + s]; __syncthreads(); }
    if (c == 0) s_dst[n * 4 + hh] = red[t];
}

// order-preserving float <-> uint mapping for atomic max
__device__ __forceinline__ unsigned f2ord(float f) {
    unsigned u = __float_as_uint(f);
    return (u & 0x80000000u) ? ~u : (u | 0x80000000u);
}
__device__ __forceinline__ float ord2f(unsigned u) {
    return __uint_as_float((u & 0x80000000u) ? (u ^ 0x80000000u) : ~u);
}

__global__ void init_layer_kernel(float* __restrict__ hnew, unsigned* __restrict__ amax,
                                  float* __restrict__ denom, int N)
{
    int i = blockIdx.x * blockDim.x + threadIdx.x;
    if (i < N * GHC) hnew[i] = 0.f;
    if (i < N * 4) { amax[i] = f2ord(-1e30f); denom[i] = 0.f; }
}

__global__ __launch_bounds__(256)
void edge_max_kernel(const int* __restrict__ ei, const float* __restrict__ s_src,
                     const float* __restrict__ s_dst, unsigned* __restrict__ amax,
                     int E, int N)
{
    int e = blockIdx.x * blockDim.x + threadIdx.x;
    if (e >= E + N) return;
    int s = (e < E) ? ei[e] : (e - E);
    int d = (e < E) ? ei[E + e] : (e - E);
#pragma unroll
    for (int h = 0; h < 4; ++h) {
        float a = s_src[s * 4 + h] + s_dst[d * 4 + h];
        a = (a > 0.f) ? a : a * NEG_SLOPE;             // leaky relu
        atomicMax(&amax[d * 4 + h], f2ord(a));
    }
}

__global__ __launch_bounds__(256)
void edge_expsum_kernel(const int* __restrict__ ei, const float* __restrict__ s_src,
                        const float* __restrict__ s_dst, const unsigned* __restrict__ amax,
                        float* __restrict__ ex, float* __restrict__ denom, int E, int N)
{
    int e = blockIdx.x * blockDim.x + threadIdx.x;
    if (e >= E + N) return;
    int s = (e < E) ? ei[e] : (e - E);
    int d = (e < E) ? ei[E + e] : (e - E);
#pragma unroll
    for (int h = 0; h < 4; ++h) {
        float a = s_src[s * 4 + h] + s_dst[d * 4 + h];
        a = (a > 0.f) ? a : a * NEG_SLOPE;
        float v = expf(a - ord2f(amax[d * 4 + h]));
        ex[(size_t)e * 4 + h] = v;
        atomicAdd(&denom[d * 4 + h], v);
    }
}

// 64 threads per edge; float4 gather from hl[src], 4 f32 atomic adds into hnew[dst].
__global__ __launch_bounds__(256)
void message_kernel(const int* __restrict__ ei, const float* __restrict__ hl,
                    const float* __restrict__ ex, const float* __restrict__ denom,
                    float* __restrict__ hnew, int E, int N)
{
    long long gid = (long long)blockIdx.x * 256 + threadIdx.x;
    long long eidx = gid >> 6;
    int t = (int)(gid & 63);
    if (eidx >= (long long)E + N) return;
    int e = (int)eidx;
    int s = (e < E) ? ei[e] : (e - E);
    int d = (e < E) ? ei[E + e] : (e - E);
    int h = t >> 4;
    float coef = ex[(size_t)e * 4 + h] / fmaxf(denom[d * 4 + h], 1e-16f);
    const float4 hv = *(const float4*)&hl[(size_t)s * GHC + t * 4];
    float* o = &hnew[(size_t)d * GHC + t * 4];
    atomicAdd(o + 0, hv.x * coef);
    atomicAdd(o + 1, hv.y * coef);
    atomicAdd(o + 2, hv.z * coef);
    atomicAdd(o + 3, hv.w * coef);
}

// Per-node gating: cw = sigmoid((h_new+bias)@conf_w + conf_b) * sigmoid(conf[n]);
// h_out = (residual ? h_old : 0) + (h_new+bias)*cw ; optional relu.
__global__ __launch_bounds__(256)
void gate_update_kernel(const float* __restrict__ hnew, const float* __restrict__ gbias,
                        const float* __restrict__ conf_w, const float* __restrict__ conf_b,
                        const float* __restrict__ confidence, const float* __restrict__ h_old,
                        float* __restrict__ h_out, int residual, int do_relu, int N)
{
    __shared__ float red[256];
    int n = blockIdx.x;
    if (n >= N) return;
    int t = threadIdx.x;
    float hn = hnew[(size_t)n * GHC + t] + gbias[t];
    red[t] = hn * conf_w[t];
    __syncthreads();
    for (int s = 128; s > 0; s >>= 1) { if (t < s) red[t] += red[t + s]; __syncthreads(); }
    float dot = red[0];
    float cw = (1.f / (1.f + expf(-(dot + conf_b[0])))) *
               (1.f / (1.f + expf(-confidence[n])));
    float v = hn * cw;
    if (residual) v += h_old[(size_t)n * GHC + t];
    if (do_relu) v = fmaxf(v, 0.f);
    h_out[(size_t)n * GHC + t] = v;
}

// Tiny second MLP layer: out[n, 0..outd) = (sig?)(hidden[n,:]@w2 + b2)
__global__ __launch_bounds__(256)
void head_kernel(const float* __restrict__ hidden, const float* __restrict__ w2,
                 const float* __restrict__ b2, float* __restrict__ out,
                 int N, int outd, int do_sig)
{
    int n = blockIdx.x * blockDim.x + threadIdx.x;
    if (n >= N) return;
    for (int j = 0; j < outd; ++j) {
        float acc = b2[j];
        for (int c = 0; c < 64; ++c) acc += hidden[(size_t)n * 64 + c] * w2[c * outd + j];
        if (do_sig) acc = 1.f / (1.f + expf(-acc));
        out[(size_t)n * outd + j] = acc;
    }
}

extern "C" void kernel_launch(void* const* d_in, const int* in_sizes, int n_in,
                              void* d_out, int out_size, void* d_ws, size_t ws_size,
                              hipStream_t stream)
{
    const int N = GN, E = GE;
    const float* x          = (const float*)d_in[0];
    const int*   ei         = (const int*)d_in[1];
    const float* confidence = (const float*)d_in[2];
    const float* w_in       = (const float*)d_in[3];
    const float* b_in       = (const float*)d_in[4];
    const float* lin_w0     = (const float*)d_in[5];
    const float* lin_w12    = (const float*)d_in[6];
    const float* att_src    = (const float*)d_in[7];
    const float* att_dst    = (const float*)d_in[8];
    const float* gat_bias   = (const float*)d_in[9];
    const float* conf_w     = (const float*)d_in[10];
    const float* conf_b     = (const float*)d_in[11];
    const float* nc_w1 = (const float*)d_in[12]; const float* nc_b1 = (const float*)d_in[13];
    const float* nc_w2 = (const float*)d_in[14]; const float* nc_b2 = (const float*)d_in[15];
    const float* oc_w1 = (const float*)d_in[16]; const float* oc_b1 = (const float*)d_in[17];
    const float* oc_w2 = (const float*)d_in[18]; const float* oc_b2 = (const float*)d_in[19];
    const float* ec_w1 = (const float*)d_in[20]; const float* ec_b1 = (const float*)d_in[21];
    const float* ec_w2 = (const float*)d_in[22]; const float* ec_b2 = (const float*)d_in[23];

    // workspace carve-out (floats)
    float* ws = (float*)d_ws;
    size_t off = 0;
    float*    buf_h  = ws + off; off += (size_t)N * GHC;
    float*    buf_hl = ws + off; off += (size_t)N * GHC;
    float*    buf_hn = ws + off; off += (size_t)N * GHC;
    float*    h0     = ws + off; off += (size_t)N * GC;
    float*    s_src  = ws + off; off += (size_t)N * 4;
    float*    s_dst  = ws + off; off += (size_t)N * 4;
    unsigned* amax   = (unsigned*)(ws + off); off += (size_t)N * 4;
    float*    denom  = ws + off; off += (size_t)N * 4;
    float*    ex     = ws + off; off += (size_t)(E + N) * 4;
    float*    hidden = ws + off; off += (size_t)N * GC;

    float* out        = (float*)d_out;
    float* node_pred  = out;
    float* origin_pred= out + N;
    float* etype_pred = out + 2 * N;
    float* out_h      = out + 6 * N;   // final h lives directly in d_out

    dim3 blk(256);
    dim3 gemm_grid_256(GHC / 32, (N + 63) / 64);   // M=256
    dim3 gemm_grid_64(GC / 32, (N + 63) / 64);     // M=64
    int edge_blocks = (E + N + 255) / 256;
    long long msg_threads = (long long)(E + N) * 64;
    int msg_blocks = (int)((msg_threads + 255) / 256);
    int init_blocks = (N * GHC + 255) / 256;

    // 1) input projection: h0 = relu(x @ w_in + b_in)   [20000x256 @ 256x64]
    gemm_wmma_kernel<<<gemm_grid_64, blk, 0, stream>>>(x, w_in, b_in, h0, N, GIN, GC, 1);

    // 2) three GAT layers
    for (int i = 0; i < 3; ++i) {
        const float* Ain = (i == 0) ? h0 : buf_h;
        int K            = (i == 0) ? GC : GHC;
        const float* W   = (i == 0) ? lin_w0 : (lin_w12 + (size_t)(i - 1) * GHC * GHC);

        gemm_wmma_kernel<<<gemm_grid_256, blk, 0, stream>>>(Ain, W, nullptr, buf_hl, N, K, GHC, 0);
        attn_score_kernel<<<N, blk, 0, stream>>>(buf_hl, att_src + i * GHC, att_dst + i * GHC,
                                                 s_src, s_dst, N);
        init_layer_kernel<<<init_blocks, blk, 0, stream>>>(buf_hn, amax, denom, N);
        edge_max_kernel<<<edge_blocks, blk, 0, stream>>>(ei, s_src, s_dst, amax, E, N);
        edge_expsum_kernel<<<edge_blocks, blk, 0, stream>>>(ei, s_src, s_dst, amax, ex, denom, E, N);
        message_kernel<<<msg_blocks, blk, 0, stream>>>(ei, buf_hl, ex, denom, buf_hn, E, N);

        float* h_out = (i == 2) ? out_h : buf_h;
        gate_update_kernel<<<N, blk, 0, stream>>>(buf_hn, gat_bias + i * GHC, conf_w, conf_b,
                                                  confidence, buf_h, h_out,
                                                  /*residual=*/(i > 0) ? 1 : 0,
                                                  /*relu=*/(i < 2) ? 1 : 0, N);
    }

    // 3) output heads (hidden reused sequentially)
    int head_blocks = (N + 255) / 256;
    gemm_wmma_kernel<<<gemm_grid_64, blk, 0, stream>>>(out_h, nc_w1, nc_b1, hidden, N, GHC, GC, 1);
    head_kernel<<<head_blocks, blk, 0, stream>>>(hidden, nc_w2, nc_b2, node_pred, N, 1, 1);

    gemm_wmma_kernel<<<gemm_grid_64, blk, 0, stream>>>(out_h, oc_w1, oc_b1, hidden, N, GHC, GC, 1);
    head_kernel<<<head_blocks, blk, 0, stream>>>(hidden, oc_w2, oc_b2, origin_pred, N, 1, 1);

    gemm_wmma_kernel<<<gemm_grid_64, blk, 0, stream>>>(out_h, ec_w1, ec_b1, hidden, N, GHC, GC, 1);
    head_kernel<<<head_blocks, blk, 0, stream>>>(hidden, ec_w2, ec_b2, etype_pred, N, 4, 0);
}
